// DeepHEATNet_80324478369832
// MI455X (gfx1250) — matrix operand order repaired and checked by reading, not compiled, not run on hardware
//
#include <hip/hip_runtime.h>

typedef __attribute__((ext_vector_type(16))) _Float16 v16h;
typedef __attribute__((ext_vector_type(8)))  float    v8f;

#define SLOPE 0.2f

// ---------------------------------------------------------------------------
// WMMA GEMM: out[M,Nn] = act( A[M,K] @ W[K,Nn](ldw) + bias )
// A fp32 row-major (lda=K), W fp32 row-major, out fp32 row-major (ldo=Nn).
// K must be a multiple of 32 (all uses: 64/128/256). NT = number of 16-wide
// N tiles (Npad = NT*16). One wave computes a 16 x (NT*16) strip: one A
// fragment feeds NT v_wmma_f32_16x16x32_f16 per K-step. All NT B fragments
// are loaded from LDS before the WMMA burst so only one s_wait_dscnt gates
// the 4 back-to-back WMMAs. act: 0=none, 1=leaky_relu(0.2).
// ---------------------------------------------------------------------------
template <int NT>
__global__ __launch_bounds__(256) void wmma_gemm_kernel(
    const float* __restrict__ A, int M, int K,
    const float* __restrict__ W, int ldw, int Nn,
    const float* __restrict__ bias, float* __restrict__ out, int act)
{
    __shared__ __align__(32) _Float16 WT[16384];   // [NT*16][K] f16, <=32KB
    const int Npad = NT * 16;

    // cooperative stage: WT[col*K + k] = W[k*ldw + col] (zero-pad cols >= Nn)
    int total = Npad * K;
    for (int idx = threadIdx.x; idx < total; idx += 256) {
        int col = idx / K;
        int k   = idx - col * K;
        WT[idx] = (col < Nn) ? (_Float16)W[(size_t)k * ldw + col] : (_Float16)0.0f;
    }
    __syncthreads();

    int tilesM = (M + 15) >> 4;
    int tileM  = blockIdx.x * 8 + (threadIdx.x >> 5);
    if (tileM >= tilesM) return;

    int lane = threadIdx.x & 31;
    // A fragment (16-bit A 16x32): lanes 0-15: M=lane,    K = 0..15
    //                              lanes 16-31: M=lane-16, K = 16..31
    int rowA = tileM * 16 + (lane & 15);
    int rowC = rowA < M ? rowA : (M - 1);   // clamp OOB rows; mask at store
    int kbA  = (lane >> 4) * 16;
    // B fragment (32x16): lanes 0-15 hold K=0..15, lanes 16-31 hold K=16..31
    int colB = lane & 15;
    int kbB  = (lane >> 4) * 16;

    const float4* ap = (const float4*)(A + (size_t)rowC * K + kbA);

    v8f acc[NT];
#pragma unroll
    for (int t = 0; t < NT; ++t) acc[t] = (v8f){};

    for (int k0 = 0; k0 < K; k0 += 32) {
        int q = k0 >> 2;                 // float4 index within the row
        float4 x0 = ap[q + 0];           // K = k0+kbA+0..3
        float4 x1 = ap[q + 1];           // K = k0+kbA+4..7
        float4 x2 = ap[q + 2];           // K = k0+kbA+8..11
        float4 x3 = ap[q + 3];           // K = k0+kbA+12..15
        if (k0 + 32 < K) __builtin_prefetch(ap + q + 8, 0, 0);

        // stage all B fragments first (one dscnt wait, then WMMA burst)
        v16h b[NT];
#pragma unroll
        for (int t = 0; t < NT; ++t)
            b[t] = *(const v16h*)&WT[(t * 16 + colB) * K + k0 + kbB];

        v16h a;
        a[0]  = (_Float16)x0.x; a[1]  = (_Float16)x0.y;
        a[2]  = (_Float16)x0.z; a[3]  = (_Float16)x0.w;
        a[4]  = (_Float16)x1.x; a[5]  = (_Float16)x1.y;
        a[6]  = (_Float16)x1.z; a[7]  = (_Float16)x1.w;
        a[8]  = (_Float16)x2.x; a[9]  = (_Float16)x2.y;
        a[10] = (_Float16)x2.z; a[11] = (_Float16)x2.w;
        a[12] = (_Float16)x3.x; a[13] = (_Float16)x3.y;
        a[14] = (_Float16)x3.z; a[15] = (_Float16)x3.w;

#pragma unroll
        for (int t = 0; t < NT; ++t) {
            acc[t] = __builtin_amdgcn_wmma_f32_16x16x32_f16(
                         false, a, false, b[t], (short)0, acc[t], false, false);
        }
    }

    // C/D layout: element i -> M = i + 8*(lane>=16), N = lane&15
    int rowBase = tileM * 16 + ((lane >> 4) << 3);
#pragma unroll
    for (int t = 0; t < NT; ++t) {
        int colOut = t * 16 + (lane & 15);
        if (colOut < Nn) {
            float bv = bias ? bias[colOut] : 0.0f;
#pragma unroll
            for (int i = 0; i < 8; ++i) {
                int row = rowBase + i;
                if (row < M) {
                    float v = acc[t][i] + bv;
                    if (act) v = (v >= 0.0f) ? v : SLOPE * v;
                    out[(size_t)row * Nn + colOut] = v;
                }
            }
        }
    }
}

static void launch_gemm(const float* A, int M, int K, const float* W, int ldw,
                        int Nn, const float* bias, float* out, int act,
                        hipStream_t s)
{
    int tilesM = (M + 15) / 16;
    int blocks = (tilesM + 7) / 8;
    if (Nn > 16) {
        wmma_gemm_kernel<4><<<blocks, 256, 0, s>>>(A, M, K, W, ldw, Nn, bias, out, act);
    } else {
        wmma_gemm_kernel<1><<<blocks, 256, 0, s>>>(A, M, K, W, ldw, Nn, bias, out, act);
    }
}

// ---------------------------------------------------------------------------
// ete table: T[t,h] = sum_k leaky(ete_emb[t,k]) * att_w[128+k, h]  (8x4)
// ---------------------------------------------------------------------------
__global__ void ete_table_kernel(const float* __restrict__ ete_emb,
                                 const float* __restrict__ att_w,
                                 float* __restrict__ T)
{
    int i = threadIdx.x;
    if (i >= 32) return;
    int t = i >> 2, h = i & 3;
    float acc = 0.0f;
    for (int k = 0; k < 16; ++k) {
        float ev = ete_emb[t * 16 + k];
        ev = (ev >= 0.0f) ? ev : SLOPE * ev;
        acc += ev * att_w[(128 + k) * 4 + h];
    }
    T[i] = acc;
}

// order-preserving float<->uint for atomic segment max
__device__ __forceinline__ unsigned f2ord(float f) {
    unsigned u = __float_as_uint(f);
    return (u & 0x80000000u) ? ~u : (u | 0x80000000u);
}
__device__ __forceinline__ float ord2f(unsigned u) {
    unsigned v = (u & 0x80000000u) ? (u & 0x7fffffffu) : ~u;
    return __uint_as_float(v);
}

// alpha_e,h = leaky(Ai[dst]+Aj[src]+T[type]+Aea[e]); scatter segment max
__global__ __launch_bounds__(256) void edge_alpha_max_kernel(
    const int* __restrict__ src, const int* __restrict__ dst,
    const int* __restrict__ etype,
    const float* __restrict__ Ai4, const float* __restrict__ Aj4,
    const float* __restrict__ Aea, const float* __restrict__ T,
    float* __restrict__ alpha, unsigned* __restrict__ amaxU, int E)
{
    int e = blockIdx.x * 256 + threadIdx.x;
    if (e >= E) return;
    int s = src[e], d = dst[e], t = etype[e];
#pragma unroll
    for (int h = 0; h < 4; ++h) {
        float v = Ai4[d * 4 + h] + Aj4[s * 4 + h] + T[t * 4 + h]
                + Aea[(size_t)e * 4 + h];
        v = (v >= 0.0f) ? v : SLOPE * v;
        alpha[(size_t)e * 4 + h] = v;
        atomicMax(&amaxU[d * 4 + h], f2ord(v));
    }
}

// w = exp(alpha - amax[dst]); scatter-add denominators
__global__ __launch_bounds__(256) void edge_exp_sum_kernel(
    const int* __restrict__ dst, float* __restrict__ alpha,
    const unsigned* __restrict__ amaxU, float* __restrict__ denom, int E)
{
    int e = blockIdx.x * 256 + threadIdx.x;
    if (e >= E) return;
    int d = dst[e];
#pragma unroll
    for (int h = 0; h < 4; ++h) {
        float a = alpha[(size_t)e * 4 + h];
        float m = ord2f(amaxU[d * 4 + h]);
        float w = __expf(a - m);
        alpha[(size_t)e * 4 + h] = w;
        atomicAdd(&denom[d * 4 + h], w);
    }
}

// out[dst, h*64+c] += (Xl[src,c] + Mea[e,c]) * w / (denom[dst,h] + 1e-16)
__global__ __launch_bounds__(256) void edge_aggregate_kernel(
    const int* __restrict__ src, const int* __restrict__ dst,
    const float* __restrict__ Xl, const float* __restrict__ Mea,
    const float* __restrict__ alpha, const float* __restrict__ denom,
    float* __restrict__ out, long long EC)
{
    long long gid = (long long)blockIdx.x * 256 + threadIdx.x;
    if (gid >= EC) return;
    int c = (int)(gid & 63);
    int e = (int)(gid >> 6);
    int s = src[e], d = dst[e];
    float val = Xl[(size_t)s * 64 + c] + Mea[(size_t)e * 64 + c];
#pragma unroll
    for (int h = 0; h < 4; ++h) {
        float a = alpha[(size_t)e * 4 + h] / (denom[d * 4 + h] + 1e-16f);
        atomicAdd(&out[(size_t)d * 256 + h * 64 + c], val * a);
    }
}

// batch norm (population stats over rows) + relu, in place; one block per col
__global__ __launch_bounds__(256) void bn_relu_kernel(
    float* __restrict__ H, const float* __restrict__ gamma,
    const float* __restrict__ beta, int rows, int cols)
{
    int col = blockIdx.x;
    int tid = threadIdx.x;
    __shared__ float s1[256];
    __shared__ float s2[256];
    float sum = 0.0f, sq = 0.0f;
    for (int r = tid; r < rows; r += 256) {
        float v = H[(size_t)r * cols + col];
        sum += v; sq += v * v;
    }
    s1[tid] = sum; s2[tid] = sq;
    __syncthreads();
    for (int s = 128; s > 0; s >>= 1) {
        if (tid < s) { s1[tid] += s1[tid + s]; s2[tid] += s2[tid + s]; }
        __syncthreads();
    }
    float mu  = s1[0] / rows;
    float var = s2[0] / rows - mu * mu;
    float rs  = rsqrtf(var + 1e-5f);
    float g = gamma[col], b = beta[col];
    for (int r = tid; r < rows; r += 256) {
        size_t i = (size_t)r * cols + col;
        float v = (H[i] - mu) * rs * g + b;
        H[i] = v > 0.0f ? v : 0.0f;
    }
}

// ---------------------------------------------------------------------------
// Orchestration. Input order (setup_inputs dict order, params flattened in
// insertion order): 0:x 1:edge_index 2:edge_types 3:edge_attr 4:llm_w 5:llm_b
// 6+7l: hl_w hl_b ete_emb ea_w att_w lin_w lin_b (l=0..2), 27..30: bn gamma/beta.
// att_w rows: [0:64)=x_i, [64:128)=x_j, [128:144)=ete, [144:208)=ea.
// lin_w rows: [0:64)=x_j part, [64:128)=ea part.
// ---------------------------------------------------------------------------
extern "C" void kernel_launch(void* const* d_in, const int* in_sizes, int n_in,
                              void* d_out, int out_size, void* d_ws, size_t ws_size,
                              hipStream_t stream)
{
    const int IN_CH = 256;
    const int N = in_sizes[0] / IN_CH;   // 50000
    const int E = in_sizes[2];           // 800000

    const float* x     = (const float*)d_in[0];
    const int*   eidx  = (const int*)d_in[1];
    const int*   etype = (const int*)d_in[2];
    const float* eattr = (const float*)d_in[3];
    const float* llm_w = (const float*)d_in[4];
    const float* llm_b = (const float*)d_in[5];
    const int* srcI = eidx;
    const int* dstI = eidx + E;

    float* ws = (float*)d_ws;
    float* Xproj = ws;                 ws += (size_t)N * 64;
    float* H     = ws;                 ws += (size_t)N * 256;
    float* Xp    = ws;                 ws += (size_t)N * 64;
    float* Xl    = ws;                 ws += (size_t)N * 64;
    float* Ai4   = ws;                 ws += (size_t)N * 4;
    float* Aj4   = ws;                 ws += (size_t)N * 4;
    unsigned* amaxU = (unsigned*)ws;   ws += (size_t)N * 4;
    float* denom = ws;                 ws += (size_t)N * 4;
    float* Tbuf  = ws;                 ws += 32;
    float* alpha = ws;                 ws += (size_t)E * 4;
    float* Aea   = ws;                 ws += (size_t)E * 4;
    float* EAproj= ws;                 ws += (size_t)E * 64;
    float* EAct  = ws;                 ws += (size_t)E * 64;
    float* Mea   = ws;                 ws += (size_t)E * 64;

    // initial_projection on nodes and edge attributes
    launch_gemm(x,     N, IN_CH, llm_w, 64, 64, llm_b, Xproj,  0, stream);
    launch_gemm(eattr, E, IN_CH, llm_w, 64, 64, llm_b, EAproj, 0, stream);

    int eBlocks = (E + 255) / 256;
    long long EC = (long long)E * 64;
    int ecBlocks = (int)((EC + 255) / 256);

    for (int l = 0; l < 3; ++l) {
        const float* hl_w  = (const float*)d_in[6 + l * 7 + 0];
        const float* hl_b  = (const float*)d_in[6 + l * 7 + 1];
        const float* ete   = (const float*)d_in[6 + l * 7 + 2];
        const float* ea_w  = (const float*)d_in[6 + l * 7 + 3];
        const float* att_w = (const float*)d_in[6 + l * 7 + 4];
        const float* lin_w = (const float*)d_in[6 + l * 7 + 5];
        const float* lin_b = (const float*)d_in[6 + l * 7 + 6];

        const float* Xin = (l == 0) ? Xproj : H;
        int Kin = (l == 0) ? 64 : 256;

        // node hetero-linear, edge-attr embedding (leaky)
        launch_gemm(Xin,    N, Kin, hl_w, 64, 64, hl_b,    Xp,   0, stream);
        launch_gemm(EAproj, E, 64,  ea_w, 64, 64, nullptr, EAct, 1, stream);
        // attention contributions (gathers pushed past the linear)
        launch_gemm(Xp,   N, 64, att_w,           4, 4, nullptr, Ai4, 0, stream);
        launch_gemm(Xp,   N, 64, att_w +  64 * 4, 4, 4, nullptr, Aj4, 0, stream);
        launch_gemm(EAct, E, 64, att_w + 144 * 4, 4, 4, nullptr, Aea, 0, stream);
        ete_table_kernel<<<1, 32, 0, stream>>>(ete, att_w, Tbuf);

        hipMemsetAsync(amaxU, 0, (size_t)N * 4 * sizeof(unsigned), stream);
        hipMemsetAsync(denom, 0, (size_t)N * 4 * sizeof(float), stream);
        edge_alpha_max_kernel<<<eBlocks, 256, 0, stream>>>(
            srcI, dstI, etype, Ai4, Aj4, Aea, Tbuf, alpha, amaxU, E);
        edge_exp_sum_kernel<<<eBlocks, 256, 0, stream>>>(
            dstI, alpha, amaxU, denom, E);

        // message linear split: node part (with bias) + edge part
        launch_gemm(Xp,   N, 64, lin_w,           64, 64, lin_b,   Xl,  0, stream);
        launch_gemm(EAct, E, 64, lin_w + 64 * 64, 64, 64, nullptr, Mea, 0, stream);

        float* target = (l == 2) ? (float*)d_out : H;
        hipMemsetAsync(target, 0, (size_t)N * 256 * sizeof(float), stream);
        edge_aggregate_kernel<<<ecBlocks, 256, 0, stream>>>(
            srcI, dstI, Xl, Mea, alpha, denom, target, EC);

        if (l < 2) {
            const float* g = (const float*)d_in[27 + l * 2];
            const float* b = (const float*)d_in[28 + l * 2];
            bn_relu_kernel<<<256, 256, 0, stream>>>(H, g, b, N, 256);
        }
    }
}
